// IndividualSpeakerIdentifier_83184926589228
// MI455X (gfx1250) — compile-verified, matrix operand order checked by reading, not run on hardware
//
#include <hip/hip_runtime.h>
#include <hip/hip_fp16.h>

typedef __attribute__((ext_vector_type(16))) _Float16 v16h;
typedef __attribute__((ext_vector_type(8)))  float    v8f;

#define B_   16
#define P_   512
#define D_   1024
#define W_   400
#define T_   32
#define E_   64
#define H_   600
#define NPAD 608     // 600 padded to 38 tiles of 16
#define K4   4096    // 4*D

// ---------------- k0: init output with bias b1 ----------------
__global__ void k_init_out(float* __restrict__ out, const float* __restrict__ b1, int n) {
    int i = blockIdx.x * blockDim.x + threadIdx.x;
    if (i < n) out[i] = b1[i & 1];
}

// ---------------- k1: wordpiece -> word segment means ----------------
__global__ void k_word_mean(const float* __restrict__ piece, const int* __restrict__ widx,
                            float* __restrict__ word) {
    int bw = blockIdx.x;                 // b*W + w
    int b  = bw / W_;
    int s  = widx[bw * 2 + 0];
    int e  = widx[bw * 2 + 1];
    float inv = 1.0f / (float)(e - s);
    int d = threadIdx.x * 4;             // 256 threads * 4 = 1024
    const float* base = piece + (size_t)b * P_ * D_ + d;
    float4 acc = make_float4(0.f, 0.f, 0.f, 0.f);
    for (int p = s; p < e; ++p) {
        float4 v = *(const float4*)(base + (size_t)p * D_);
        acc.x += v.x; acc.y += v.y; acc.z += v.z; acc.w += v.w;
    }
    acc.x *= inv; acc.y *= inv; acc.z *= inv; acc.w *= inv;
    *(float4*)(word + (size_t)bw * D_ + d) = acc;
}

// ---------------- k2: word -> trigger/entity span means (packed f16) ----------------
__global__ void k_span_mean(const float* __restrict__ word, const int* __restrict__ tidx,
                            const int* __restrict__ eidx, _Float16* __restrict__ trig_h,
                            _Float16* __restrict__ ent_h) {
    int id = blockIdx.x;                 // [0, B*T) triggers, then entities
    int b, s, e;
    _Float16* dsth;
    if (id < B_ * T_) {
        b = id / T_;
        s = tidx[id * 2 + 0]; e = tidx[id * 2 + 1];
        dsth = trig_h + (size_t)id * D_;
    } else {
        int id2 = id - B_ * T_;
        b = id2 / E_;
        s = eidx[id2 * 2 + 0]; e = eidx[id2 * 2 + 1];
        dsth = ent_h + (size_t)id2 * D_;
    }
    float inv = 1.0f / (float)(e - s);
    int d = threadIdx.x * 4;
    const float* base = word + (size_t)b * W_ * D_ + d;
    float4 acc = make_float4(0.f, 0.f, 0.f, 0.f);
    for (int w = s; w < e; ++w) {
        float4 v = *(const float4*)(base + (size_t)w * D_);
        acc.x += v.x; acc.y += v.y; acc.z += v.z; acc.w += v.w;
    }
    union { _Float16 h[4]; uint2 u; } r;
    r.h[0] = (_Float16)(acc.x * inv); r.h[1] = (_Float16)(acc.y * inv);
    r.h[2] = (_Float16)(acc.z * inv); r.h[3] = (_Float16)(acc.w * inv);
    *(uint2*)(dsth + d) = r.u;
}

// ---------------- k3: W0 [4096,600] f32 -> W0^T [608,4096] f16 (zero pad) ----------------
__global__ void k_w0_transpose(const float* __restrict__ W0, _Float16* __restrict__ w0t) {
    int idx = blockIdx.x * blockDim.x + threadIdx.x;
    if (idx >= NPAD * K4) return;
    int n = idx / K4;
    int k = idx - n * K4;
    float v = (n < H_) ? W0[(size_t)k * H_ + n] : 0.0f;
    w0t[idx] = (_Float16)v;
}

// ---------------- k4: per-span precontribution GEMM: C = A @ W0seg (+ b0) ----------------
// A: [M][1024] f16 row-major; w0t column n contiguous over k; koff selects segment.
// grid.x = M/64, grid.y = NPAD/32; 8 waves = 4(M) x 2(N).
__launch_bounds__(256, 2)
__global__ void k_pre_gemm(const _Float16* __restrict__ A, const _Float16* __restrict__ w0t,
                           int koff, const float* __restrict__ b0, int addBias,
                           float* __restrict__ C) {
    const int tid  = threadIdx.x;
    const int wave = tid >> 5, lane = tid & 31;
    const int lh = lane >> 4, lm = lane & 15;
    const int m0 = blockIdx.x * 64 + (wave & 3) * 16;
    const int n  = blockIdx.y * 32 + (wave >> 2) * 16 + lm;
    const _Float16* arow = A   + (size_t)(m0 + lm) * D_ + 8 * lh;
    const _Float16* brow = w0t + (size_t)n * K4 + koff + 8 * lh;
    v8f acc = {};
    #pragma unroll 4
    for (int kk = 0; kk < D_; kk += 32) {
        union { uint4 u[2]; v16h v; } ua, ub;
        ua.u[0] = *(const uint4*)(arow + kk);
        ua.u[1] = *(const uint4*)(arow + kk + 16);
        ub.u[0] = *(const uint4*)(brow + kk);
        ub.u[1] = *(const uint4*)(brow + kk + 16);
        acc = __builtin_amdgcn_wmma_f32_16x16x32_f16(false, ua.v, false, ub.v,
                                                     (short)0, acc, false, false);
    }
    float bias = (addBias && n < H_) ? b0[n] : 0.f;
    #pragma unroll
    for (int j = 0; j < 8; ++j)
        C[(size_t)(m0 + j + 8 * lh) * NPAD + n] = acc[j] + bias;
}

// ---------------- k5: fused (t*e, |t-e|) WMMA GEMM, all-register A construction --------
// grid.x = B*T (one (b,t); 64 rows = all entities), grid.y = NPAD/32.
// A fragments built in registers from f16 span reps with packed f16 VALU ops:
// no LDS, no barriers in the K loop.
__launch_bounds__(256, 2)
__global__ void k_role_ffn(const _Float16* __restrict__ trig_h,
                           const _Float16* __restrict__ ent_h,
                           const _Float16* __restrict__ w0t, const float* __restrict__ tc,
                           const float* __restrict__ ec, const float* __restrict__ w1,
                           float* __restrict__ out) {
    const int tid = threadIdx.x;
    const int bt  = blockIdx.x;             // b*T + t
    const int b   = bt / T_;

    const int wave = tid >> 5;
    const int lane = tid & 31;
    const int lh   = lane >> 4;
    const int lm   = lane & 15;
    const int m0   = (wave & 3) * 16;       // entity-row tile base
    const int n    = blockIdx.y * 32 + (wave >> 2) * 16 + lm;

    const _Float16* erow  = ent_h  + ((size_t)b * E_ + m0 + lm) * D_ + 8 * lh;
    const _Float16* trow  = trig_h + (size_t)bt * D_ + 8 * lh;
    const _Float16* browC = w0t + (size_t)n * K4 + 2 * D_ + 8 * lh;  // t*e   segment
    const _Float16* browD = w0t + (size_t)n * K4 + 3 * D_ + 8 * lh;  // |t-e| segment

    v8f acc = {};

    #pragma unroll 2
    for (int kk = 0; kk < D_; kk += 32) {
        union { uint4 u[2]; v16h v; unsigned s[8]; } ae, at, ubC, ubD, aq;
        ae.u[0] = *(const uint4*)(erow + kk);
        ae.u[1] = *(const uint4*)(erow + kk + 16);
        at.u[0] = *(const uint4*)(trow + kk);
        at.u[1] = *(const uint4*)(trow + kk + 16);
        ubC.u[0] = *(const uint4*)(browC + kk);
        ubC.u[1] = *(const uint4*)(browC + kk + 16);
        ubD.u[0] = *(const uint4*)(browD + kk);
        ubD.u[1] = *(const uint4*)(browD + kk + 16);

        v16h prod = at.v * ae.v;            // v_pk_mul_f16
        aq.v = at.v - ae.v;                 // v_pk_sub_f16
        #pragma unroll
        for (int j = 0; j < 8; ++j) aq.s[j] &= 0x7fff7fffu;   // packed |.|

        acc = __builtin_amdgcn_wmma_f32_16x16x32_f16(false, prod, false, ubC.v,
                                                     (short)0, acc, false, false);
        acc = __builtin_amdgcn_wmma_f32_16x16x32_f16(false, aq.v, false, ubD.v,
                                                     (short)0, acc, false, false);
    }

    // epilogue: h = relu(c + tc[bt][n] + ec[b*64+e][n]); out[row] += h * W1[n][:]
    float w10 = 0.f, w11 = 0.f;
    if (n < H_) { w10 = w1[n * 2 + 0]; w11 = w1[n * 2 + 1]; }
    const float tcv = tc[(size_t)bt * NPAD + n];    // includes b0
    #pragma unroll
    for (int j = 0; j < 8; ++j) {
        const int erow_idx = m0 + j + 8 * lh;       // entity index = row within block
        float h = acc[j] + tcv + ec[((size_t)b * E_ + erow_idx) * NPAD + n];
        h = (n < H_) ? (h > 0.f ? h : 0.f) : 0.f;
        float p0 = h * w10;
        float p1 = h * w11;
        #pragma unroll
        for (int sft = 8; sft >= 1; sft >>= 1) {
            p0 += __shfl_xor(p0, sft, 16);
            p1 += __shfl_xor(p1, sft, 16);
        }
        if (lm == 0) {
            int row = bt * 64 + erow_idx;
            atomicAdd(out + row * 2 + 0, p0);
            atomicAdd(out + row * 2 + 1, p1);
        }
    }
}

extern "C" void kernel_launch(void* const* d_in, const int* in_sizes, int n_in,
                              void* d_out, int out_size, void* d_ws, size_t ws_size,
                              hipStream_t stream) {
    const float* piece = (const float*)d_in[0];
    const int*   widx  = (const int*)  d_in[1];
    const int*   tidx  = (const int*)  d_in[2];
    const int*   eidx  = (const int*)  d_in[3];
    const float* W0    = (const float*)d_in[4];
    const float* b0    = (const float*)d_in[5];
    const float* W1    = (const float*)d_in[6];
    const float* b1    = (const float*)d_in[7];
    float* out = (float*)d_out;

    // workspace carve-up
    float* word = (float*)d_ws;                              // 16*400*1024 f32
    float* tc   = word + (size_t)B_ * W_ * D_;               // 512*608  f32 (trig@W0a + b0)
    float* ec   = tc   + (size_t)B_ * T_ * NPAD;             // 1024*608 f32 (ent@W0b)
    _Float16* trig_h = (_Float16*)(ec + (size_t)B_ * E_ * NPAD);  // 512*1024  f16
    _Float16* ent_h  = trig_h + (size_t)B_ * T_ * D_;             // 1024*1024 f16
    _Float16* w0t    = ent_h  + (size_t)B_ * E_ * D_;             // 608*4096  f16

    // k0: out = bias b1
    k_init_out<<<(out_size + 255) / 256, 256, 0, stream>>>(out, b1, out_size);

    // k1: word means
    k_word_mean<<<B_ * W_, 256, 0, stream>>>(piece, widx, word);

    // k2: trigger + entity span means (f16)
    k_span_mean<<<B_ * (T_ + E_), 256, 0, stream>>>(word, tidx, eidx, trig_h, ent_h);

    // k3: W0 -> f16 transposed, padded
    k_w0_transpose<<<(NPAD * K4 + 255) / 256, 256, 0, stream>>>(W0, w0t);

    // k4a: tc = trig @ W0[k=0:1024] + b0
    {
        dim3 g((B_ * T_) / 64, NPAD / 32);
        k_pre_gemm<<<g, 256, 0, stream>>>(trig_h, w0t, 0, b0, 1, tc);
    }
    // k4b: ec = ent @ W0[k=1024:2048]
    {
        dim3 g((B_ * E_) / 64, NPAD / 32);
        k_pre_gemm<<<g, 256, 0, stream>>>(ent_h, w0t, D_, b0, 0, ec);
    }

    // k5: fused (t*e, |t-e|) WMMA GEMM + combine + relu + second layer
    {
        dim3 g(B_ * T_, NPAD / 32);
        k_role_ffn<<<g, 256, 0, stream>>>(trig_h, ent_h, w0t, tc, ec, W1, out);
    }
}
